// GRU_27659589386769
// MI455X (gfx1250) — compile-verified, matrix operand order checked by reading, not run on hardware
//
#include <hip/hip_runtime.h>
#include <math.h>

// ---------------- problem constants ----------------
constexpr int HDIM = 256;     // hidden / feature dim
constexpr int BB   = 64;      // batch
constexpr int SS   = 2048;    // sequence length
constexpr int MTOT = BB * SS; // 131072 rows for input projection GEMMs
constexpr int NWG  = 16;      // persistent workgroups in the scan kernel

// ---------------- WMMA types ----------------
typedef __attribute__((ext_vector_type(16))) __bf16 bf16x16;
typedef __attribute__((ext_vector_type(8)))  float  f32x8;

struct F4 { float x, y, z, w; };
struct U4 { unsigned int x, y, z, w; };

union Frag {
  bf16x16        v;
  unsigned short h[16];
  U4             q[2];
};

__device__ __forceinline__ unsigned short f2bf(float f) {
  unsigned int u = __float_as_uint(f);
  u += 0x7FFFu + ((u >> 16) & 1u);          // round-to-nearest-even
  return (unsigned short)(u >> 16);
}
__device__ __forceinline__ float bf2f(unsigned short s) {
  return __uint_as_float(((unsigned int)s) << 16);
}

// Convert one lane-chunk (16 elements, two 8-float runs) of a 16x32 A-fragment.
__device__ __forceinline__ void cvtA_bf16(Frag& a, const float* __restrict__ rowp) {
  const F4* p0 = reinterpret_cast<const F4*>(rowp);
  F4 f0 = p0[0], f1 = p0[1];
  const F4* p1 = reinterpret_cast<const F4*>(rowp + 16);
  F4 f2 = p1[0], f3 = p1[1];
  a.h[0]  = f2bf(f0.x); a.h[1]  = f2bf(f0.y); a.h[2]  = f2bf(f0.z); a.h[3]  = f2bf(f0.w);
  a.h[4]  = f2bf(f1.x); a.h[5]  = f2bf(f1.y); a.h[6]  = f2bf(f1.z); a.h[7]  = f2bf(f1.w);
  a.h[8]  = f2bf(f2.x); a.h[9]  = f2bf(f2.y); a.h[10] = f2bf(f2.z); a.h[11] = f2bf(f2.w);
  a.h[12] = f2bf(f3.x); a.h[13] = f2bf(f3.y); a.h[14] = f2bf(f3.z); a.h[15] = f2bf(f3.w);
}

// Packed A/B fragment flat index helpers (both use [tile][kstep][lane][e] chunks).
// A layout per 16x32 tile: lane L holds row L%16, kbase=(L/16)*8,
//   K(e) = kstep*32 + (e<8 ? kbase+e : 16+kbase+e-8).

// ---------------- kernel 1: pack 6 weight matrices into bf16 B-fragment layout ------
// Per matrix: [ntile=16][kstep=8][lane=32][e=16] bf16 = 65536 elements.
__global__ __launch_bounds__(256) void pack_weights(
    const float* __restrict__ Wi_r, const float* __restrict__ Wi_z, const float* __restrict__ Wi_n,
    const float* __restrict__ Wh_r, const float* __restrict__ Wh_z, const float* __restrict__ Wh_n,
    unsigned short* __restrict__ packW) {
  int idx   = blockIdx.x * 256 + threadIdx.x;   // 0 .. 6*65536-1
  int mat   = idx >> 16;
  int r     = idx & 0xFFFF;
  int e     = r & 15;
  int lane  = (r >> 4) & 31;
  int kstep = (r >> 9) & 7;
  int ntile = (r >> 12) & 15;
  int n     = ntile * 16 + (lane & 15);
  int kbase = (lane >> 4) << 3;
  int k     = kstep * 32 + (e < 8 ? kbase + e : 16 + kbase + (e - 8));
  const float* W = (mat == 0) ? Wi_r : (mat == 1) ? Wi_z : (mat == 2) ? Wi_n
                 : (mat == 3) ? Wh_r : (mat == 4) ? Wh_z : Wh_n;
  packW[idx] = f2bf(W[k * HDIM + n]);
}

// ---------------- kernel 2: pack x into bf16 A-fragment layout ----------------
// xpack: [mtile=8192][kstep=8][lane=32][e=16].  One thread = one 16-elem chunk.
__global__ __launch_bounds__(256) void pack_x(
    const float* __restrict__ x, unsigned short* __restrict__ xpack) {
  const int t     = blockIdx.x * 256 + threadIdx.x;   // 0 .. 8192*8*32-1
  const int lane  = t & 31;
  const int kstep = (t >> 5) & 7;
  const int mtile = t >> 8;
  const int m     = mtile * 16 + (lane & 15);
  const int kbase = (lane >> 4) << 3;
  Frag a;
  cvtA_bf16(a, x + (long)m * HDIM + kstep * 32 + kbase);
  U4* dst = reinterpret_cast<U4*>(xpack) + (long)t * 2;
  dst[0] = a.q[0];
  dst[1] = a.q[1];
}

// ---------------- kernel 3: zero h0 (fp32 + packed bf16) and barrier counter --------
__global__ __launch_bounds__(256) void init_state(float* __restrict__ hbuf0,
                                                  unsigned int* __restrict__ hpack0,
                                                  unsigned int* __restrict__ cnt) {
  int i = blockIdx.x * 256 + threadIdx.x;
  if (i < BB * HDIM) {
    hbuf0[i]  = 0.0f;
    if (i < (BB * HDIM) / 2) hpack0[i] = 0u;   // 16384 bf16 = 8192 dwords
  }
  if (i == 0) *cnt = 0u;
}

// ---------------- kernel 4: input projections (3 GEMMs, bf16 WMMA) ----------------
// One wave computes one 16x16 output tile for all 3 weight matrices (A reuse x3).
// Inner loop is pure b128 loads + WMMA (A pre-packed by pack_x).
__global__ __launch_bounds__(256) void input_proj(
    const unsigned short* __restrict__ xpack, const unsigned short* __restrict__ packW,
    unsigned short* __restrict__ projR, unsigned short* __restrict__ projZ,
    unsigned short* __restrict__ projN) {
  const int tid   = threadIdx.x;
  const int lane  = tid & 31;
  const int wave  = tid >> 5;
  const int tile  = blockIdx.x * 8 + wave;   // 0 .. 131071  (8192 mtiles x 16 ntiles)
  const int mtile = tile >> 4;
  const int ntile = tile & 15;

  const f32x8 z8 = {0.f, 0.f, 0.f, 0.f, 0.f, 0.f, 0.f, 0.f};
  f32x8 acc[3] = {z8, z8, z8};

#pragma unroll
  for (int ks = 0; ks < 8; ++ks) {
    Frag a;
    const U4* ap = reinterpret_cast<const U4*>(xpack) + ((long)(mtile * 8 + ks) * 32 + lane) * 2;
    a.q[0] = ap[0]; a.q[1] = ap[1];
    const long bidx = ((long)(ntile * 8 + ks) * 32 + lane) * 2;
#pragma unroll
    for (int g = 0; g < 3; ++g) {
      Frag b;
      const U4* wp = reinterpret_cast<const U4*>(packW) + (long)g * 8192 + bidx;
      b.q[0] = wp[0]; b.q[1] = wp[1];
      acc[g] = __builtin_amdgcn_wmma_f32_16x16x32_bf16(
          false, a.v, false, b.v, (short)0, acc[g], false, false);
    }
  }

  // C layout: lane L, vgpr v -> M = v + 8*(L/16), N = L%16
  const int rowbase = mtile * 16 + ((lane >> 4) << 3);
  const int col     = ntile * 16 + (lane & 15);
#pragma unroll
  for (int v = 0; v < 8; ++v) {
    const long o = (long)(rowbase + v) * HDIM + col;
    projR[o] = f2bf(acc[0][v]);
    projZ[o] = f2bf(acc[1][v]);
    projN[o] = f2bf(acc[2][v]);
  }
}

// ---------------- kernel 5: persistent recurrent scan ----------------
// 16 WGs x 384 threads (12 waves). WG wg owns H-columns [wg*16, wg*16+16).
// Wave w: gate = w/4 (r,z,n), mtile = w%4. A operands come from a packed bf16
// double buffer written by the previous step's elementwise stage, so the
// critical loop is b128 loads + WMMA only. Grid barrier = device-scope atomics.
__global__ __launch_bounds__(384) void gru_scan(
    const unsigned short* __restrict__ packW,
    const unsigned short* __restrict__ projR, const unsigned short* __restrict__ projZ,
    const unsigned short* __restrict__ projN,
    const float* __restrict__ b_r, const float* __restrict__ b_z, const float* __restrict__ b_n,
    float* __restrict__ hbuf,            // fp32 double buffer: 2 x 64 x 256
    unsigned short* __restrict__ hpack,  // bf16 A-frag double buffer: 2 x 16384
    unsigned int* __restrict__ cnt,
    float* __restrict__ out) {           // [B=64][S=2048][H=256] fp32
  const int tid   = threadIdx.x;
  const int lane  = tid & 31;
  const int wave  = tid >> 5;       // 0..11
  const int gate  = wave >> 2;      // 0..2
  const int mtile = wave & 3;       // 0..3
  const int wg    = blockIdx.x;     // 0..15 (= ntile)

  __shared__ float lds[3][64][16];

  const unsigned short* whg = packW + (long)(3 + gate) * 65536;
  const int colbase = wg * 16;
  const f32x8 z8 = {0.f, 0.f, 0.f, 0.f, 0.f, 0.f, 0.f, 0.f};

  for (int s = 0; s < SS; ++s) {
    const float*          hcur  = hbuf  + (size_t)(s & 1) * (BB * HDIM);
    float*                hnxt  = hbuf  + (size_t)((s + 1) & 1) * (BB * HDIM);
    const unsigned short* hpcur = hpack + (size_t)(s & 1) * (BB * HDIM);
    unsigned short*       hpnxt = hpack + (size_t)((s + 1) & 1) * (BB * HDIM);

    // ---- GEMM: this wave's 16x16 tile of (h @ Wh_gate), K = 256 ----
    f32x8 acc = z8;
#pragma unroll
    for (int ks = 0; ks < 8; ++ks) {
      Frag a;
      const U4* ap = reinterpret_cast<const U4*>(hpcur) + ((long)(mtile * 8 + ks) * 32 + lane) * 2;
      a.q[0] = ap[0]; a.q[1] = ap[1];
      Frag b;
      const U4* wp = reinterpret_cast<const U4*>(whg) + ((long)(wg * 8 + ks) * 32 + lane) * 2;
      b.q[0] = wp[0]; b.q[1] = wp[1];
      acc = __builtin_amdgcn_wmma_f32_16x16x32_bf16(
          false, a.v, false, b.v, (short)0, acc, false, false);
    }
    {
      const int mloc = mtile * 16 + ((lane >> 4) << 3);
      const int n    = lane & 15;
#pragma unroll
      for (int v = 0; v < 8; ++v) lds[gate][mloc + v][n] = acc[v];
    }
    __syncthreads();

    // ---- elementwise gate math over this WG's 64x16 slice ----
    if (tid < 256) {
      const int m  = tid >> 2;          // 0..63 (batch row)
      const int nb = (tid & 3) * 4;     // 0,4,8,12
      // prefetch next timestep's projection lines (global_prefetch_b8)
      if (s + 1 < SS && nb == 0) {
        const long pnext = ((long)m * SS + (s + 1)) * HDIM + colbase;
        __builtin_prefetch(projR + pnext, 0, 3);
        __builtin_prefetch(projZ + pnext, 0, 3);
        __builtin_prefetch(projN + pnext, 0, 3);
      }
#pragma unroll
      for (int j = 0; j < 4; ++j) {
        const int n  = nb + j;
        const int nc = colbase + n;
        const long pidx = ((long)m * SS + s) * HDIM + nc;
        const float xr = bf2f(projR[pidx]);
        const float xz = bf2f(projZ[pidx]);
        const float xn = bf2f(projN[pidx]);
        const float hold = hcur[m * HDIM + nc];
        const float r  = 1.0f / (1.0f + __expf(-(xr + lds[0][m][n] + b_r[nc])));
        const float z  = 1.0f / (1.0f + __expf(-(xz + lds[1][m][n] + b_z[nc])));
        const float nn = tanhf(xn + r * (lds[2][m][n] + b_n[nc]));
        const float hnew = (1.0f - z) * nn + z * hold;
        hnxt[m * HDIM + nc] = hnew;
        out[pidx] = hnew;
        // scatter h' into next step's packed bf16 A-fragment layout:
        // element (m, k=nc): mtile=m>>4, rm=m&15, kstep=k>>5, kk=k&31,
        // half=(kk>>3)&1, lane=rm+16*half, e=(kk&7)+((kk&16)>>1)
        {
          const int kk   = nc & 31;
          const int dstl = (m & 15) + ((kk >> 3) & 1) * 16;
          const int e    = (kk & 7) + ((kk & 16) >> 1);
          const long d   = ((long)((m >> 4) * 8 + (nc >> 5)) * 32 + dstl) * 16 + e;
          hpnxt[d] = f2bf(hnew);
        }
      }
    }

    // ---- device-scope grid barrier (release add, acquire spin) ----
    __threadfence();
    __syncthreads();
    if (tid == 0) {
      __hip_atomic_fetch_add(cnt, 1u, __ATOMIC_RELEASE, __HIP_MEMORY_SCOPE_AGENT);
      const unsigned int target = (unsigned int)(NWG * (s + 1));
      while (__hip_atomic_load(cnt, __ATOMIC_ACQUIRE, __HIP_MEMORY_SCOPE_AGENT) < target)
        __builtin_amdgcn_s_sleep(2);
    }
    __syncthreads();
    __threadfence();
  }
}

// ---------------- host entry ----------------
extern "C" void kernel_launch(void* const* d_in, const int* in_sizes, int n_in,
                              void* d_out, int out_size, void* d_ws, size_t ws_size,
                              hipStream_t stream) {
  (void)in_sizes; (void)n_in; (void)out_size; (void)ws_size;

  const float* x    = (const float*)d_in[0];
  const float* Wi_r = (const float*)d_in[1];
  const float* Wi_z = (const float*)d_in[2];
  const float* Wi_n = (const float*)d_in[3];
  const float* Wh_r = (const float*)d_in[4];
  const float* Wh_z = (const float*)d_in[5];
  const float* Wh_n = (const float*)d_in[6];
  const float* b_r  = (const float*)d_in[7];
  const float* b_z  = (const float*)d_in[8];
  const float* b_n  = (const float*)d_in[9];

  // workspace layout (bytes)
  char* ws = (char*)d_ws;
  unsigned short* packW = (unsigned short*)ws;                 // 6*65536*2   = 786432
  unsigned int*   cnt   = (unsigned int*)(ws + 786432);        // 128 reserved
  float*          hbuf  = (float*)(ws + 786560);               // 2*16384*4   = 131072
  unsigned short* hpack = (unsigned short*)(ws + 917632);      // 2*16384*2   = 65536
  unsigned short* xpack = (unsigned short*)(ws + 983168);      // 131072*256*2 = 64 MiB
  unsigned short* projR = (unsigned short*)(ws + 983168 + (size_t)MTOT * HDIM * 2);
  unsigned short* projZ = projR + (size_t)MTOT * HDIM;
  unsigned short* projN = projZ + (size_t)MTOT * HDIM;

  pack_weights<<<(6 * 65536) / 256, 256, 0, stream>>>(Wi_r, Wi_z, Wi_n, Wh_r, Wh_z, Wh_n, packW);
  pack_x<<<(MTOT / 16) * 8 * 32 / 256, 256, 0, stream>>>(x, xpack);
  init_state<<<(BB * HDIM + 255) / 256, 256, 0, stream>>>(hbuf, (unsigned int*)hpack, cnt);
  input_proj<<<MTOT * 16 / 16 / 8, 256, 0, stream>>>(xpack, packW, projR, projZ, projN);
  gru_scan<<<NWG, 384, 0, stream>>>(packW, projR, projZ, projN, b_r, b_z, b_n,
                                    hbuf, hpack, cnt, (float*)d_out);
}
